// WorkingMemoryModule_82119774699678
// MI455X (gfx1250) — matrix operand order repaired and checked by reading, not compiled, not run on hardware
//
#include <hip/hip_runtime.h>
#include <hip/hip_bf16.h>
#include <math.h>

// Problem constants (match reference): B=256, M=512, D=768, H=8, C=768, dh=96
#define B_  256
#define M_  512
#define D_  768
#define H_  8
#define C_  768
#define DH_ 96

typedef __attribute__((ext_vector_type(16))) __bf16 v16bf;
typedef __attribute__((ext_vector_type(8)))  float  v8f;
typedef __attribute__((ext_vector_type(4)))  float  f4;

__device__ __forceinline__ void cvt4(v16bf& dst, int base, f4 v)
{
    dst[base + 0] = (__bf16)v.x;
    dst[base + 1] = (__bf16)v.y;
    dst[base + 2] = (__bf16)v.z;
    dst[base + 3] = (__bf16)v.w;
}

__device__ __forceinline__ float fast_sigmoid(float x)
{
    // native v_rcp_f32 instead of the IEEE div expansion
    return __builtin_amdgcn_rcpf(1.0f + __expf(-x));
}

// ---------------------------------------------------------------------------
// Generic WMMA GEMM:  out[16m x 16n tile] = act( scale * (A @ W^T) + bias )
// A: row-major (rows x K), row stride lda.  W: row-major (N rows x K cols),
// so B[k][n] = W[n][k]  ->  A @ W^T.  Per-blockIdx.z pointer strides allow
// per-head batching. One wave (32 threads) per 16x16 tile.
// All fragment loads are 128-bit (every lda / K offset here is a multiple
// of 4 floats and buffers are 16B aligned).
// ---------------------------------------------------------------------------
__global__ void __launch_bounds__(32)
gemm_awt(const float* __restrict__ A, int lda, long long aZ,
         const float* __restrict__ W, int ldw, long long wZ,
         const float* __restrict__ bias,
         float* __restrict__ out, int ldo, long long oZ,
         int K, int act, float scale)
{
    const int lane = threadIdx.x;
    const int l16  = lane & 15;
    const int hi   = lane >> 4;
    const int m0   = blockIdx.x * 16;
    const int n0   = blockIdx.y * 16;
    const long long z = blockIdx.z;
    A   += z * aZ;
    W   += z * wZ;
    out += z * oZ;

    const int kbA = hi ? 8 : 0;    // A-frag: K offset for lane half
    const int kbB = hi ? 8 : 0;    // B-frag: K offset for lane half
    const float* arowp = A + (long long)(m0 + l16) * lda;
    const float* wrowp = W + (long long)(n0 + l16) * ldw;

    v8f acc = {};
    #pragma unroll 2
    for (int k0 = 0; k0 < K; k0 += 32) {
        const f4* pa = (const f4*)(arowp + k0 + kbA);        // k, k+4
        const f4* pa2 = (const f4*)(arowp + k0 + 16 + kbA);  // k+16, k+20
        const f4* pw = (const f4*)(wrowp + k0 + kbB);        // k, k+4
        const f4* pw2 = (const f4*)(wrowp + k0 + 16 + kbB);  // k+16, k+20
        f4 a0 = pa[0],  a1 = pa[1];
        f4 a2 = pa2[0], a3 = pa2[1];
        f4 b0 = pw[0], b1 = pw[1], b2 = pw2[0], b3 = pw2[1];

        v16bf a, b;
        cvt4(a, 0, a0);  cvt4(a, 4, a1);  cvt4(a, 8, a2);  cvt4(a, 12, a3);
        cvt4(b, 0, b0);  cvt4(b, 4, b1);  cvt4(b, 8, b2);  cvt4(b, 12, b3);

        acc = __builtin_amdgcn_wmma_f32_16x16x32_bf16(
            false, a, false, b, (short)0, acc, false, false);
    }

    const float bv = bias ? bias[n0 + l16] : 0.0f;
    #pragma unroll
    for (int r = 0; r < 8; ++r) {
        float v = acc[r] * scale + bv;
        if (act == 1) v = fast_sigmoid(v);
        out[(long long)(m0 + r + 8 * hi) * ldo + n0 + l16] = v;
    }
}

// ---------------------------------------------------------------------------
// WMMA GEMM with K-major B:  out = A @ Bm, Bm[k][n] at Bm[k*ldb + n].
// Used for ctx = attn @ V (per head via blockIdx.z strides). B fragment is
// a column walk (stride ldb), so those stay scalar loads; A is 128-bit.
// ---------------------------------------------------------------------------
__global__ void __launch_bounds__(32)
gemm_ab(const float* __restrict__ A, int lda, long long aZ,
        const float* __restrict__ Bm, int ldb, long long bZ,
        float* __restrict__ out, int ldo, long long oZ, int K)
{
    const int lane = threadIdx.x;
    const int l16  = lane & 15;
    const int hi   = lane >> 4;
    const int m0   = blockIdx.x * 16;
    const int n0   = blockIdx.y * 16;
    const long long z = blockIdx.z;
    A   += z * aZ;
    Bm  += z * bZ;
    out += z * oZ;

    const int kbA = hi ? 8 : 0;
    const int kbB = hi ? 8 : 0;
    const float* arowp = A + (long long)(m0 + l16) * lda;

    v8f acc = {};
    #pragma unroll 2
    for (int k0 = 0; k0 < K; k0 += 32) {
        const f4* pa = (const f4*)(arowp + k0 + kbA);
        const f4* pa2 = (const f4*)(arowp + k0 + 16 + kbA);
        f4 a0 = pa[0],  a1 = pa[1];
        f4 a2 = pa2[0], a3 = pa2[1];

        v16bf a, b;
        cvt4(a, 0, a0);  cvt4(a, 4, a1);  cvt4(a, 8, a2);  cvt4(a, 12, a3);

        const float* pb = Bm + (long long)(k0 + kbB) * ldb + n0 + l16;
        #pragma unroll
        for (int j = 0; j < 16; ++j) b[j] = (__bf16)pb[(long long)(j < 8 ? j : j + 8) * ldb];

        acc = __builtin_amdgcn_wmma_f32_16x16x32_bf16(
            false, a, false, b, (short)0, acc, false, false);
    }
    #pragma unroll
    for (int r = 0; r < 8; ++r)
        out[(long long)(m0 + r + 8 * hi) * ldo + n0 + l16] = acc[r];
}

// ---------------------------------------------------------------------------
// Softmax over rows of length M_=512. One wave per row (16 elems/lane).
// blockIdx.y selects the read-scores or write-scores array.
// ---------------------------------------------------------------------------
__global__ void __launch_bounds__(32)
softmax_rows(float* __restrict__ sc_r, float* __restrict__ sc_w)
{
    float* p = (blockIdx.y == 0 ? sc_r : sc_w) + (long long)blockIdx.x * M_;
    const int lane = threadIdx.x;
    float v[16];
    float mx = -INFINITY;
    #pragma unroll
    for (int i = 0; i < 16; ++i) { v[i] = p[lane + 32 * i]; mx = fmaxf(mx, v[i]); }
    #pragma unroll
    for (int off = 16; off > 0; off >>= 1) mx = fmaxf(mx, __shfl_xor(mx, off));
    float s = 0.0f;
    #pragma unroll
    for (int i = 0; i < 16; ++i) { v[i] = __expf(v[i] - mx); s += v[i]; }
    #pragma unroll
    for (int off = 16; off > 0; off >>= 1) s += __shfl_xor(s, off);
    const float inv = __builtin_amdgcn_rcpf(s);
    #pragma unroll
    for (int i = 0; i < 16; ++i) p[lane + 32 * i] = v[i] * inv;
}

// ww[b,m] = mean over heads of write attention probs (layout [B][H][M])
__global__ void __launch_bounds__(256)
mean_heads(const float* __restrict__ attn_w, float* __restrict__ ww)
{
    const int idx = blockIdx.x * 256 + threadIdx.x;   // 0 .. B*M-1
    const int b = idx >> 9;          // /512
    const int m = idx & (M_ - 1);
    float s = 0.0f;
    #pragma unroll
    for (int h = 0; h < H_; ++h)
        s += attn_w[((long long)(b * H_ + h)) * M_ + m];
    ww[idx] = s * (1.0f / H_);
}

// ug[b], fg[b] = sigmoid([cs, read_data] . gate_w + gate_b)
__global__ void __launch_bounds__(256)
gates_kernel(const float* __restrict__ cs, const float* __restrict__ rd,
             const float* __restrict__ ugw, const float* __restrict__ ugb,
             const float* __restrict__ fgw, const float* __restrict__ fgb,
             float* __restrict__ ug, float* __restrict__ fg)
{
    const int b = blockIdx.x, t = threadIdx.x;
    float su = 0.0f, sf = 0.0f;
    for (int i = t; i < C_; i += 256) {
        const float x = cs[(long long)b * C_ + i];
        su += x * ugw[i];
        sf += x * fgw[i];
    }
    for (int i = t; i < D_; i += 256) {
        const float x = rd[(long long)b * D_ + i];
        su += x * ugw[C_ + i];
        sf += x * fgw[C_ + i];
    }
    __shared__ float shu[256], shf[256];
    shu[t] = su; shf[t] = sf;
    __syncthreads();
    for (int s = 128; s > 0; s >>= 1) {
        if (t < s) { shu[t] += shu[t + s]; shf[t] += shf[t + s]; }
        __syncthreads();
    }
    if (t == 0) {
        ug[b] = fast_sigmoid(shu[0] + ugb[0]);
        fg[b] = fast_sigmoid(shf[0] + fgb[0]);
    }
}

// ---------------------------------------------------------------------------
// The 403 MB streaming kernel (runtime-dominant):
// upd[b,m,d] = mem[m,d] + ww[b,m]*(wd[b,d]*ug[b] - ev[b,d]*fg[b])
// 128-bit NT stores so the giant output stream doesn't evict the L2-resident
// memory/ww/per-batch rows.
// ---------------------------------------------------------------------------
__global__ void __launch_bounds__(192)
update_mem(const float* __restrict__ memory, const float* __restrict__ wd,
           const float* __restrict__ ev, const float* __restrict__ ww,
           const float* __restrict__ ug, const float* __restrict__ fg,
           float* __restrict__ upd)
{
    const int m = blockIdx.x;
    const int b = blockIdx.y;
    const int d = threadIdx.x * 4;
    const float w = ww[(long long)b * M_ + m];
    const float u = w * ug[b];
    const float f = w * fg[b];
    const f4 m4 = *(const f4*)(memory + (long long)m * D_ + d);
    const f4 w4 = *(const f4*)(wd + (long long)b * D_ + d);
    const f4 e4 = *(const f4*)(ev + (long long)b * D_ + d);
    f4 o;
    o.x = m4.x + (w4.x * u - e4.x * f);
    o.y = m4.y + (w4.y * u - e4.y * f);
    o.z = m4.z + (w4.z * u - e4.z * f);
    o.w = m4.w + (w4.w * u - e4.w * f);
    f4* dst = (f4*)(upd + ((long long)b * M_ + m) * D_ + d);
    __builtin_nontemporal_store(o, dst);
}

// ---------------------------------------------------------------------------
extern "C" void kernel_launch(void* const* d_in, const int* in_sizes, int n_in,
                              void* d_out, int out_size, void* d_ws, size_t ws_size,
                              hipStream_t stream)
{
    const float* cs      = (const float*)d_in[0];   // (B,C)
    const float* wdata   = (const float*)d_in[1];   // (B,D)
    const float* memory  = (const float*)d_in[2];   // (M,D)
    const float* in_w    = (const float*)d_in[3];   // (3D,D): wq | wk | wv
    const float* in_b    = (const float*)d_in[4];   // (3D,)
    const float* out_w   = (const float*)d_in[5];   // (D,D)
    const float* out_b   = (const float*)d_in[6];
    const float* read_w  = (const float*)d_in[7];   // (D,C)
    const float* read_b  = (const float*)d_in[8];
    const float* write_w = (const float*)d_in[9];
    const float* write_b = (const float*)d_in[10];
    const float* erase_w = (const float*)d_in[11];
    const float* erase_b = (const float*)d_in[12];
    const float* ug_w    = (const float*)d_in[13];  // (1, C+D)
    const float* ug_b    = (const float*)d_in[14];
    const float* fg_w    = (const float*)d_in[15];
    const float* fg_b    = (const float*)d_in[16];

    float* outp      = (float*)d_out;
    float* read_data = outp;                            // (B,D)
    float* upd       = outp + (size_t)B_ * D_;          // (B,M,D)

    // Workspace carve-up (~16.8 MB of floats)
    float* ws   = (float*)d_ws;
    const size_t BD  = (size_t)B_ * D_;
    const size_t MD  = (size_t)M_ * D_;
    const size_t BHM = (size_t)B_ * H_ * M_;
    float* rk   = ws;            // read key              (B,D)
    float* wkey = rk   + BD;     // write key             (B,D)
    float* ev   = wkey + BD;     // erase vec (sigmoid)   (B,D)
    float* qr   = ev   + BD;     // read query            (B,D)
    float* qw   = qr   + BD;     // write query           (B,D)
    float* ctx  = qw   + BD;     // attention context     (B,D)
    float* Kmat = ctx  + BD;     // K projection          (M,D)
    float* Vmat = Kmat + MD;     // V projection          (M,D)
    float* sc_r = Vmat + MD;     // read scores/probs     (B,H,M)
    float* sc_w = sc_r + BHM;    // write scores/probs    (B,H,M)
    float* wwv  = sc_w + BHM;    // mean write weights    (B,M)
    float* ugv  = wwv  + (size_t)B_ * M_;  // (B,)
    float* fgv  = ugv  + B_;               // (B,)

    const dim3 w32(32);
    const float attn_scale = 1.0f / sqrtf((float)DH_);

    // Key / erase projections from controller state (B x C) @ (D x C)^T
    gemm_awt<<<dim3(B_/16, D_/16, 1), w32, 0, stream>>>(
        cs, C_, 0, read_w,  C_, 0, read_b,  rk,   D_, 0, C_, 0, 1.0f);
    gemm_awt<<<dim3(B_/16, D_/16, 1), w32, 0, stream>>>(
        cs, C_, 0, write_w, C_, 0, write_b, wkey, D_, 0, C_, 0, 1.0f);
    gemm_awt<<<dim3(B_/16, D_/16, 1), w32, 0, stream>>>(
        cs, C_, 0, erase_w, C_, 0, erase_b, ev,   D_, 0, C_, 1, 1.0f);

    // K/V computed ONCE (memory is shared across the batch)
    gemm_awt<<<dim3(M_/16, D_/16, 1), w32, 0, stream>>>(
        memory, D_, 0, in_w + (size_t)D_*D_,   D_, 0, in_b + D_,   Kmat, D_, 0, D_, 0, 1.0f);
    gemm_awt<<<dim3(M_/16, D_/16, 1), w32, 0, stream>>>(
        memory, D_, 0, in_w + (size_t)2*D_*D_, D_, 0, in_b + 2*D_, Vmat, D_, 0, D_, 0, 1.0f);

    // Query projections: q = key @ wq^T + bq
    gemm_awt<<<dim3(B_/16, D_/16, 1), w32, 0, stream>>>(
        rk,   D_, 0, in_w, D_, 0, in_b, qr, D_, 0, D_, 0, 1.0f);
    gemm_awt<<<dim3(B_/16, D_/16, 1), w32, 0, stream>>>(
        wkey, D_, 0, in_w, D_, 0, in_b, qw, D_, 0, D_, 0, 1.0f);

    // Per-head scores (B x M per head), z = head: scores = q_h @ K_h^T * scale
    gemm_awt<<<dim3(B_/16, M_/16, H_), w32, 0, stream>>>(
        qr, D_, DH_, Kmat, D_, DH_, nullptr, sc_r, H_*M_, M_, DH_, 0, attn_scale);
    gemm_awt<<<dim3(B_/16, M_/16, H_), w32, 0, stream>>>(
        qw, D_, DH_, Kmat, D_, DH_, nullptr, sc_w, H_*M_, M_, DH_, 0, attn_scale);

    // Softmax both score sets, then head-mean of write probs -> ww
    softmax_rows<<<dim3(B_*H_, 2), w32, 0, stream>>>(sc_r, sc_w);
    mean_heads<<<dim3((B_*M_)/256), dim3(256), 0, stream>>>(sc_w, wwv);

    // ctx = attn_r @ V  (per head)
    gemm_ab<<<dim3(B_/16, DH_/16, H_), w32, 0, stream>>>(
        sc_r, H_*M_, M_, Vmat, D_, DH_, ctx, D_, DH_, M_);

    // read_data = ctx @ out_w^T + out_b  (written straight into d_out)
    gemm_awt<<<dim3(B_/16, D_/16, 1), w32, 0, stream>>>(
        ctx, D_, 0, out_w, D_, 0, out_b, read_data, D_, 0, D_, 0, 1.0f);

    // Gates from [cs, read_data]
    gates_kernel<<<dim3(B_), dim3(256), 0, stream>>>(
        cs, read_data, ug_w, ug_b, fg_w, fg_b, ugv, fgv);

    // The 403 MB streaming memory update (runtime-dominant; NT stores)
    update_mem<<<dim3(M_, B_), dim3(192), 0, stream>>>(
        memory, wdata, ev, wwv, ugv, fgv, upd);
}
